// DecoderLayer_17910013625017
// MI455X (gfx1250) — matrix operand order repaired and checked by reading, not compiled, not run on hardware
//
#include <hip/hip_runtime.h>

typedef __attribute__((ext_vector_type(16))) __bf16 bf16x16;
typedef __attribute__((ext_vector_type(8)))  float  f32x8;

static constexpr int SQ    = 2048;
static constexpr int HID   = 2048;
static constexpr int NHEAD = 16;
static constexpr int NKVH  = 8;
static constexpr int HDIM  = 128;
static constexpr int FFD   = 6144;
static constexpr float EPSV  = 1e-6f;
static constexpr float NEGV  = -1e9f;
static constexpr float SCALE = 0.08838834764831845f; // HDIM^-0.5

__device__ __forceinline__ f32x8 wmma_bf16(bf16x16 a, bf16x16 b, f32x8 c) {
  return __builtin_amdgcn_wmma_f32_16x16x32_bf16(false, a, false, b, (short)0, c, false, false);
}
// Assemble a 32-byte fragment from two 16-byte contiguous chunks (b128 loads)
__device__ __forceinline__ bf16x16 ld_frag(const __bf16* p0, const __bf16* p1) {
  union { uint4 q[2]; bf16x16 f; } u;
  u.q[0] = *(const uint4*)p0;
  u.q[1] = *(const uint4*)p1;
  return u.f;
}
// CDNA5 async memory->LDS copy (16B per lane), tracked by ASYNCcnt
__device__ __forceinline__ void async_ld_b128(void* lds_dst, const void* gsrc) {
  uint32_t l = (uint32_t)(uintptr_t)lds_dst;   // flat LDS aperture: offset in addr[31:0]
  asm volatile("global_load_async_to_lds_b128 %0, %1, off"
               :: "v"(l), "v"(gsrc) : "memory");
}
__device__ __forceinline__ void wait_async0() {
  asm volatile("s_wait_asynccnt 0x0" ::: "memory");
}

// ---------------- fp32 -> bf16 transpose convert: out[N][K] = in[K][N] ----------------
__global__ __launch_bounds__(256) void f2bfT(const float* __restrict__ in,
                                             __bf16* __restrict__ out, int K, int N) {
  __shared__ float t[32][33];
  const int tx = threadIdx.x, ty = threadIdx.y;
  const int kb = blockIdx.y * 32, nb = blockIdx.x * 32;
  #pragma unroll
  for (int j = 0; j < 4; ++j)
    t[ty + 8 * j][tx] = in[(size_t)(kb + ty + 8 * j) * N + nb + tx];
  __syncthreads();
  #pragma unroll
  for (int j = 0; j < 4; ++j)
    out[(size_t)(nb + ty + 8 * j) * K + kb + tx] = (__bf16)t[tx][ty + 8 * j];
}

// ---------------- row RMSNorm -> bf16 ----------------
__global__ __launch_bounds__(256) void rmsnorm_bf16(
    const float* __restrict__ X, const float* __restrict__ W,
    __bf16* __restrict__ Y, int ncols) {
  __shared__ float red[256];
  const int row = blockIdx.x, tid = threadIdx.x;
  const float* x = X + (size_t)row * ncols;
  float ss = 0.f;
  for (int i = tid; i < ncols; i += 256) { float v = x[i]; ss += v * v; }
  red[tid] = ss; __syncthreads();
  for (int s = 128; s > 0; s >>= 1) { if (tid < s) red[tid] += red[tid + s]; __syncthreads(); }
  const float inv = rsqrtf(red[0] / (float)ncols + EPSV);
  for (int i = tid; i < ncols; i += 256)
    Y[(size_t)row * ncols + i] = (__bf16)(x[i] * inv * W[i]);
}

// ------------- per-head RMSNorm + RoPE, repack head-major -------------
__global__ __launch_bounds__(128) void qk_norm_rope(
    const float* __restrict__ in,   // [S, nheads*HDIM]
    const float* __restrict__ nw,   // [HDIM]
    const int* __restrict__ pos,    // [S]
    __bf16* __restrict__ out,       // [nheads, S, HDIM]
    int nheads) {
  const int s = blockIdx.x, h = blockIdx.y, d = threadIdx.x;
  __shared__ float xn[HDIM];
  __shared__ float red[HDIM];
  const float v = in[((size_t)s * nheads + h) * HDIM + d];
  red[d] = v * v; __syncthreads();
  for (int st = 64; st > 0; st >>= 1) { if (d < st) red[d] += red[d + st]; __syncthreads(); }
  const float r = rsqrtf(red[0] / (float)HDIM + EPSV);
  xn[d] = v * r * nw[d];
  __syncthreads();
  const float p = (float)pos[s];
  const int dm = d & 63;
  const float fr = p * powf(1.0e6f, -(float)dm * (1.0f / 64.0f));
  const float c = cosf(fr), sn = sinf(fr);
  const float rot = (d < 64) ? -xn[d + 64] : xn[d - 64];
  out[((size_t)h * SQ + s) * HDIM + d] = (__bf16)(xn[d] * c + rot * sn);
}

// ------------- V repack head-major TRANSPOSED: out[h][d][s] -------------
__global__ __launch_bounds__(128) void v_packT(
    const float* __restrict__ in, __bf16* __restrict__ out) {
  const int s = blockIdx.x, h = blockIdx.y, d = threadIdx.x;
  out[((size_t)h * HDIM + d) * SQ + s] = (__bf16)in[((size_t)s * NKVH + h) * HDIM + d];
}

// ------- bf16 GEMM: C[M,N] = A[M,K] @ BT[N,K]^T (+Res), fp32 out -------
// 128x128 tile, BK=32, double-buffered async-to-LDS pipeline.
__global__ __launch_bounds__(256) void gemm_bf16_rtr(
    const __bf16* __restrict__ A, const __bf16* __restrict__ BT,
    const float* __restrict__ Res, float* __restrict__ C,
    int M, int N, int K) {
  __shared__ __align__(16) __bf16 As[2][128 * 32];   // [row][k]
  __shared__ __align__(16) __bf16 Bs[2][128 * 32];   // [col][k]
  const int tid = threadIdx.x;
  const int w = tid >> 5, lane = tid & 31;
  const int l16 = lane & 15, hi = lane >> 4;
  const int bm = blockIdx.y * 128, bn = blockIdx.x * 128;
  const int lrow = tid >> 2, lseg = tid & 3;   // 64 rows x 4 segs per 256-thread pass

  f32x8 acc[8] = {};

  auto issue = [&](int kt, int buf) {
    const __bf16* a0 = A + (size_t)(bm + lrow) * K + kt + lseg * 8;
    async_ld_b128(&As[buf][lrow * 32 + lseg * 8], a0);
    async_ld_b128(&As[buf][(lrow + 64) * 32 + lseg * 8], a0 + (size_t)64 * K);
    const __bf16* b0 = BT + (size_t)(bn + lrow) * K + kt + lseg * 8;
    async_ld_b128(&Bs[buf][lrow * 32 + lseg * 8], b0);
    async_ld_b128(&Bs[buf][(lrow + 64) * 32 + lseg * 8], b0 + (size_t)64 * K);
  };

  const int ktiles = K / 32;
  issue(0, 0);
  for (int it = 0; it < ktiles; ++it) {
    wait_async0();          // my async loads for tile `it` done
    __syncthreads();        // everyone's done; previous buffer fully consumed
    if (it + 1 < ktiles) issue((it + 1) * 32, (it + 1) & 1);
    const int buf = it & 1;
    const __bf16* ap = &As[buf][(w * 16 + l16) * 32 + 8 * hi];
    bf16x16 a = ld_frag(ap, ap + 16);
    // all 8 B fragments live at once -> bulk DS issue, partial waits, overlap
    bf16x16 bfr[8];
    #pragma unroll
    for (int t = 0; t < 8; ++t) {
      const __bf16* bp = &Bs[buf][(t * 16 + l16) * 32 + hi * 8];
      bfr[t] = ld_frag(bp, bp + 16);
    }
    #pragma unroll
    for (int t = 0; t < 8; ++t) acc[t] = wmma_bf16(a, bfr[t], acc[t]);
  }
  #pragma unroll
  for (int t = 0; t < 8; ++t)
    #pragma unroll
    for (int r = 0; r < 8; ++r) {
      const int gm = bm + w * 16 + r + 8 * hi;
      const int gn = bn + t * 16 + l16;
      float v = acc[t][r];
      if (Res) v += Res[(size_t)gm * N + gn];
      C[(size_t)gm * N + gn] = v;
    }
}

// ---------------- flash attention (bf16 WMMA, fp32 softmax) ----------------
__global__ __launch_bounds__(128) void flash_attn(
    const __bf16* __restrict__ Q,    // [NHEAD, S, HDIM]
    const __bf16* __restrict__ Kh,   // [NKVH, S, HDIM]
    const __bf16* __restrict__ VhT,  // [NKVH, HDIM, S]   (transposed)
    const int* __restrict__ qidx, const int* __restrict__ kidx,
    __bf16* __restrict__ Ctx)        // [S, NHEAD*HDIM]
{
  const int h = blockIdx.x, qb = blockIdx.y;
  const int kvh = h / (NHEAD / NKVH);
  const int tid = threadIdx.x, w = tid >> 5, lane = tid & 31;
  const int l16 = lane & 15, hi = lane >> 4;

  __shared__ __align__(16) __bf16 Kt[64 * HDIM];    // [key][hd]
  __shared__ __align__(16) __bf16 VtT[HDIM * 64];   // [hd][key]
  __shared__ __align__(16) float  Ps[4][16 * 64];
  __shared__ __align__(16) __bf16 Pb[4][16 * 64];
  __shared__ float rmax[4][16], rsum[4][16], rfac[4][16];
  __shared__ int Ki[64];

  const int q0 = qb * 64 + w * 16;

  bf16x16 aQ[4];
  {
    const __bf16* qrow = Q + ((size_t)h * SQ + q0 + l16) * HDIM;
    #pragma unroll
    for (int c = 0; c < 4; ++c)
      aQ[c] = ld_frag(qrow + c * 32 + 8 * hi, qrow + c * 32 + 8 * hi + 16);
  }
  int qpos[8];
  #pragma unroll
  for (int r = 0; r < 8; ++r) qpos[r] = qidx[q0 + r + 8 * hi];

  f32x8 o[8] = {};
  if (lane < 16) { rmax[w][lane] = -__builtin_inff(); rsum[w][lane] = 0.f; }
  __syncthreads();

  for (int kb = 0; kb < SQ / 64; ++kb) {
    {
      // K tile 64x128 (row-contig in hd), V^T tile 128x64 (row-contig in key)
      const __bf16* gk = Kh + ((size_t)kvh * SQ + kb * 64) * HDIM;
      const __bf16* gv = VhT + (size_t)kvh * HDIM * SQ + kb * 64;
      #pragma unroll
      for (int ii = 0; ii < 8; ++ii) {
        const int i = tid + ii * 128;
        async_ld_b128((uint4*)Kt + i, (const uint4*)gk + i);
        const int row = i >> 3, seg = i & 7;
        async_ld_b128((uint4*)VtT + i, (const uint4*)(gv + (size_t)row * SQ + seg * 8));
      }
      if (tid < 64) Ki[tid] = kidx[kb * 64 + tid];
      wait_async0();
      __syncthreads();
    }

    // S = Q @ K^T   (16x64 per wave), 4 fragments preloaded per hd-chunk
    f32x8 sc[4] = {};
    #pragma unroll
    for (int c = 0; c < 4; ++c) {
      bf16x16 bfr[4];
      #pragma unroll
      for (int t = 0; t < 4; ++t) {
        const __bf16* bp = &Kt[(t * 16 + l16) * HDIM + c * 32 + hi * 8];
        bfr[t] = ld_frag(bp, bp + 16);
      }
      #pragma unroll
      for (int t = 0; t < 4; ++t) sc[t] = wmma_bf16(aQ[c], bfr[t], sc[t]);
    }
    // scale + gathered causal mask, stage raw scores
    #pragma unroll
    for (int t = 0; t < 4; ++t)
      #pragma unroll
      for (int r = 0; r < 8; ++r) {
        const int m = r + 8 * hi, col = t * 16 + l16;
        Ps[w][m * 64 + col] = sc[t][r] * SCALE + ((Ki[col] <= qpos[r]) ? 0.f : NEGV);
      }
    __syncthreads();

    // online softmax (one row per lane 0..15)
    if (lane < 16) {
      const int m = lane;
      float mo = rmax[w][m], tmax = -__builtin_inff();
      for (int j = 0; j < 64; ++j) tmax = fmaxf(tmax, Ps[w][m * 64 + j]);
      const float mn = fmaxf(mo, tmax);
      const float f = __expf(mo - mn);
      float s = 0.f;
      for (int j = 0; j < 64; ++j) {
        const float p = __expf(Ps[w][m * 64 + j] - mn);
        Pb[w][m * 64 + j] = (__bf16)p; s += p;
      }
      rsum[w][m] = rsum[w][m] * f + s;
      rmax[w][m] = mn;
      rfac[w][m] = f;
    }
    __syncthreads();

    // rescale O then O += P @ V, 8 V fragments preloaded per K-chunk
    #pragma unroll
    for (int r = 0; r < 8; ++r) {
      const float f = rfac[w][r + 8 * hi];
      #pragma unroll
      for (int t = 0; t < 8; ++t) o[t][r] *= f;
    }
    #pragma unroll
    for (int c2 = 0; c2 < 2; ++c2) {
      const __bf16* pp = &Pb[w][l16 * 64 + c2 * 32 + 8 * hi];
      bf16x16 aP = ld_frag(pp, pp + 16);
      bf16x16 bfr[8];
      #pragma unroll
      for (int t = 0; t < 8; ++t) {
        const __bf16* bp = &VtT[(t * 16 + l16) * 64 + c2 * 32 + hi * 8];
        bfr[t] = ld_frag(bp, bp + 16);
      }
      #pragma unroll
      for (int t = 0; t < 8; ++t) o[t] = wmma_bf16(aP, bfr[t], o[t]);
    }
    __syncthreads();   // all reads of Kt/VtT/Pb done before next tile is issued
  }
  #pragma unroll
  for (int r = 0; r < 8; ++r) {
    const int m = r + 8 * hi;
    const float inv = 1.f / rsum[w][m];
    const size_t base = (size_t)(q0 + m) * (NHEAD * HDIM) + (size_t)h * HDIM;
    #pragma unroll
    for (int t = 0; t < 8; ++t)
      Ctx[base + t * 16 + l16] = (__bf16)(o[t][r] * inv);
  }
}

// ---------------- importance (last-query softmax, mean over heads) ----------------
__global__ __launch_bounds__(256) void imp_pass1(
    const __bf16* __restrict__ Q, const __bf16* __restrict__ Kh,
    float* __restrict__ logits, float* __restrict__ stats) {
  const int h = blockIdx.x, tid = threadIdx.x;
  const int kvh = h / (NHEAD / NKVH);
  __shared__ float qv[HDIM];
  __shared__ float red[256];
  if (tid < HDIM) qv[tid] = (float)Q[((size_t)h * SQ + (SQ - 1)) * HDIM + tid];
  __syncthreads();
  float lmax = -__builtin_inff();
  for (int j = tid; j < SQ; j += 256) {
    const __bf16* kr = Kh + ((size_t)kvh * SQ + j) * HDIM;
    float acc = 0.f;
    for (int d8 = 0; d8 < HDIM; d8 += 8) {
      union { uint4 q; __bf16 e[8]; } u;
      u.q = *(const uint4*)(kr + d8);
      #pragma unroll
      for (int j2 = 0; j2 < 8; ++j2) acc += qv[d8 + j2] * (float)u.e[j2];
    }
    acc *= SCALE;
    logits[(size_t)h * SQ + j] = acc;
    lmax = fmaxf(lmax, acc);
  }
  red[tid] = lmax; __syncthreads();
  for (int s = 128; s > 0; s >>= 1) { if (tid < s) red[tid] = fmaxf(red[tid], red[tid + s]); __syncthreads(); }
  const float hm = red[0]; __syncthreads();
  float ls = 0.f;
  for (int j = tid; j < SQ; j += 256) ls += __expf(logits[(size_t)h * SQ + j] - hm);
  red[tid] = ls; __syncthreads();
  for (int s = 128; s > 0; s >>= 1) { if (tid < s) red[tid] += red[tid + s]; __syncthreads(); }
  if (tid == 0) { stats[h * 2] = hm; stats[h * 2 + 1] = red[0]; }
}

__global__ void imp_pass2(const float* __restrict__ logits,
                          const float* __restrict__ stats,
                          float* __restrict__ imp) {
  const int i = blockIdx.x * blockDim.x + threadIdx.x;
  if (i >= SQ) return;
  float a = 0.f;
  for (int h = 0; h < NHEAD; ++h)
    a += __expf(logits[(size_t)h * SQ + i] - stats[h * 2]) / stats[h * 2 + 1];
  float v = a / (float)NHEAD;
  if (i == SQ - 1) v = __builtin_inff();
  imp[i] = v;
}

// ---------------- silu(gate) * up -> bf16 ----------------
__global__ void silu_mul(const float* __restrict__ g, const float* __restrict__ u,
                         __bf16* __restrict__ out, int n) {
  const int i = blockIdx.x * blockDim.x + threadIdx.x;
  if (i >= n) return;
  const float gv = g[i];
  out[i] = (__bf16)((gv / (1.f + __expf(-gv))) * u[i]);
}

// =================================================================
extern "C" void kernel_launch(void* const* d_in, const int* in_sizes, int n_in,
                              void* d_out, int out_size, void* d_ws, size_t ws_size,
                              hipStream_t stream) {
  (void)in_sizes; (void)n_in; (void)out_size; (void)ws_size;
  const float* hidden    = (const float*)d_in[0];
  /* d_in[1] causal_mask: reconstructed from idxs */
  const int*   keys_idxs = (const int*)d_in[2];
  const int*   hs_idxs   = (const int*)d_in[3];
  const int*   positions = (const int*)d_in[4];
  const float* ln_in_w   = (const float*)d_in[5];
  const float* q_w       = (const float*)d_in[6];
  const float* k_w       = (const float*)d_in[7];
  const float* v_w       = (const float*)d_in[8];
  const float* q_norm_w  = (const float*)d_in[9];
  const float* k_norm_w  = (const float*)d_in[10];
  const float* o_w       = (const float*)d_in[11];
  const float* ln_post_w = (const float*)d_in[12];
  const float* gate_w    = (const float*)d_in[13];
  const float* up_w      = (const float*)d_in[14];
  const float* down_w    = (const float*)d_in[15];
  float* out = (float*)d_out;
  float* imp = out + (size_t)SQ * HID;

  // bump allocator over d_ws
  char* base = (char*)d_ws;
  size_t off = 0;
  auto alloc = [&](size_t bytes) -> void* {
    void* r = base + off;
    off += (bytes + 255) & ~(size_t)255;
    return r;
  };
  __bf16* x_bf  = (__bf16*)alloc((size_t)SQ * HID * 2);
  __bf16* wqT   = (__bf16*)alloc((size_t)HID * NHEAD * HDIM * 2);
  __bf16* wkT   = (__bf16*)alloc((size_t)HID * NKVH * HDIM * 2);
  __bf16* wvT   = (__bf16*)alloc((size_t)HID * NKVH * HDIM * 2);
  __bf16* woT   = (__bf16*)alloc((size_t)NHEAD * HDIM * HID * 2);
  __bf16* wgT   = (__bf16*)alloc((size_t)HID * FFD * 2);
  __bf16* wuT   = (__bf16*)alloc((size_t)HID * FFD * 2);
  __bf16* wdT   = (__bf16*)alloc((size_t)FFD * HID * 2);
  float*  qf    = (float*) alloc((size_t)SQ * NHEAD * HDIM * 4);
  float*  kf    = (float*) alloc((size_t)SQ * NKVH * HDIM * 4);
  float*  vf    = (float*) alloc((size_t)SQ * NKVH * HDIM * 4);
  __bf16* qb    = (__bf16*)alloc((size_t)NHEAD * SQ * HDIM * 2);
  __bf16* kb    = (__bf16*)alloc((size_t)NKVH * SQ * HDIM * 2);
  __bf16* vbT   = (__bf16*)alloc((size_t)NKVH * HDIM * SQ * 2);
  __bf16* ctxb  = (__bf16*)alloc((size_t)SQ * NHEAD * HDIM * 2);
  float*  hres  = (float*) alloc((size_t)SQ * HID * 4);
  __bf16* yb    = (__bf16*)alloc((size_t)SQ * HID * 2);
  float*  gf    = (float*) alloc((size_t)SQ * FFD * 4);
  float*  uf    = (float*) alloc((size_t)SQ * FFD * 4);
  __bf16* mb    = (__bf16*)alloc((size_t)SQ * FFD * 2);
  float*  ilog  = (float*) alloc((size_t)NHEAD * SQ * 4);
  float*  istat = (float*) alloc((size_t)NHEAD * 2 * 4);

  // weight downconversion + transpose (fp32 [K,N] -> bf16 [N,K])
  auto convT = [&](const float* src, __bf16* dst, int K, int N) {
    f2bfT<<<dim3(N / 32, K / 32), dim3(32, 8), 0, stream>>>(src, dst, K, N);
  };
  convT(q_w,    wqT, HID, NHEAD * HDIM);
  convT(k_w,    wkT, HID, NKVH * HDIM);
  convT(v_w,    wvT, HID, NKVH * HDIM);
  convT(o_w,    woT, NHEAD * HDIM, HID);
  convT(gate_w, wgT, HID, FFD);
  convT(up_w,   wuT, HID, FFD);
  convT(down_w, wdT, FFD, HID);

  // x = rms(hidden, ln_in_w)
  rmsnorm_bf16<<<SQ, 256, 0, stream>>>(hidden, ln_in_w, x_bf, HID);

  // Q/K/V projections (WMMA GEMM)
  gemm_bf16_rtr<<<dim3((NHEAD * HDIM) / 128, SQ / 128), 256, 0, stream>>>(
      x_bf, wqT, nullptr, qf, SQ, NHEAD * HDIM, HID);
  gemm_bf16_rtr<<<dim3((NKVH * HDIM) / 128, SQ / 128), 256, 0, stream>>>(
      x_bf, wkT, nullptr, kf, SQ, NKVH * HDIM, HID);
  gemm_bf16_rtr<<<dim3((NKVH * HDIM) / 128, SQ / 128), 256, 0, stream>>>(
      x_bf, wvT, nullptr, vf, SQ, NKVH * HDIM, HID);

  // per-head RMSNorm + RoPE, repack head-major bf16
  qk_norm_rope<<<dim3(SQ, NHEAD), 128, 0, stream>>>(qf, q_norm_w, positions, qb, NHEAD);
  qk_norm_rope<<<dim3(SQ, NKVH), 128, 0, stream>>>(kf, k_norm_w, positions, kb, NKVH);
  v_packT<<<dim3(SQ, NKVH), 128, 0, stream>>>(vf, vbT);

  // importance from last query row
  imp_pass1<<<NHEAD, 256, 0, stream>>>(qb, kb, ilog, istat);
  imp_pass2<<<(SQ + 255) / 256, 256, 0, stream>>>(ilog, istat, imp);

  // flash attention
  flash_attn<<<dim3(NHEAD, SQ / 64), 128, 0, stream>>>(qb, kb, vbT, hs_idxs, keys_idxs, ctxb);

  // h = hidden + ctx @ o_w
  gemm_bf16_rtr<<<dim3(HID / 128, SQ / 128), 256, 0, stream>>>(
      ctxb, woT, hidden, hres, SQ, HID, NHEAD * HDIM);

  // y = rms(h, ln_post_w)
  rmsnorm_bf16<<<SQ, 256, 0, stream>>>(hres, ln_post_w, yb, HID);

  // MLP
  gemm_bf16_rtr<<<dim3(FFD / 128, SQ / 128), 256, 0, stream>>>(
      yb, wgT, nullptr, gf, SQ, FFD, HID);
  gemm_bf16_rtr<<<dim3(FFD / 128, SQ / 128), 256, 0, stream>>>(
      yb, wuT, nullptr, uf, SQ, FFD, HID);
  silu_mul<<<(int)(((size_t)SQ * FFD + 255) / 256), 256, 0, stream>>>(
      gf, uf, mb, SQ * FFD);
  // out = h + mlpin @ down_w
  gemm_bf16_rtr<<<dim3(HID / 128, SQ / 128), 256, 0, stream>>>(
      mb, wdT, hres, out, SQ, HID, FFD);
}
